// DiffDock_13657996001871
// MI455X (gfx1250) — matrix operand order repaired and checked by reading, not compiled, not run on hardware
//
#include <hip/hip_runtime.h>

typedef unsigned int u32;
typedef _Float16 f16;
typedef f16  v16h __attribute__((ext_vector_type(16)));
typedef float v8f  __attribute__((ext_vector_type(8)));
typedef int  vi4  __attribute__((vector_size(16)));   // native vector, not HIP_vector_type

#define NSC 16          // 0e multiplicity
#define WPB 8           // waves per block (wave32)
#define TPB 256
#define ETILE 16        // edges per wave tile
#define KPAD 64         // K padded to 2 x 32
#define W1COLS 48
#define W2COLS 320
#define SWGT_DW ((W1COLS + W2COLS) * (KPAD / 2))   // 11776 dwords = 46 KB
#define W2_OFF_DW (W1COLS * (KPAD / 2))            // 1536

// ---- native f32 atomic add (global_atomic_add_f32) with safe fallback ----
__device__ inline void atomAdd(float* p, float v) {
#if __has_builtin(__builtin_amdgcn_global_atomic_fadd_f32)
    typedef __attribute__((address_space(1))) float* gfp;
    __builtin_amdgcn_global_atomic_fadd_f32((gfp)p, v);
#else
    atomicAdd(p, v);
#endif
}

// ---- fragment loaders (layouts per CDNA5 ISA 7.12.2) ----
// A frag: 8 dwords = dwords [base..base+3] (VGPR0-3) and [base+8..base+11] (VGPR4-7)
__device__ inline v16h ldA(const u32* p) {
    union { vi4 q[2]; v16h h; } u;
    u.q[0] = *(const vi4*)(p);
    u.q[1] = *(const vi4*)(p + 8);
    return u.h;
}
// B frag: 8 dwords = dwords [base..base+3] and [base+8..base+11] (same K order as A)
__device__ inline v16h ldB(const u32* p) {
    union { vi4 q[2]; v16h h; } u;
    u.q[0] = *(const vi4*)(p);
    u.q[1] = *(const vi4*)(p + 8);
    return u.h;
}

// ---- one-time weight conversion: f32 row-major -> f16 column-major, K-pair packed, K padded to 64 ----
__global__ __launch_bounds__(256) void prep_weights(const float* __restrict__ w1,
                                                    const float* __restrict__ w2,
                                                    u32* __restrict__ swgt_g) {
    int gid = blockIdx.x * 256 + threadIdx.x;
    if (gid >= SWGT_DW) return;
    int col, kd, ld;
    const float* src;
    if (gid < W2_OFF_DW) { col = gid >> 5; kd = gid & 31; src = w1 + col; ld = 48; }
    else { int g = gid - W2_OFF_DW; col = g >> 5; kd = g & 31; src = w2 + col; ld = 320; }
    int k0 = kd * 2;
    float f0 = (k0     < 48) ? src[(size_t)k0 * ld]       : 0.0f;
    float f1 = (k0 + 1 < 48) ? src[(size_t)(k0 + 1) * ld] : 0.0f;
    union { f16 h[2]; u32 u; } p;
    p.h[0] = (f16)f0; p.h[1] = (f16)f1;
    swgt_g[gid] = p.u;
}

__global__ __launch_bounds__(256) void zero_f32(float* __restrict__ p, int n) {
    int i = blockIdx.x * 256 + threadIdx.x;
    if (i < n) p[i] = 0.0f;
}

__global__ __launch_bounds__(256) void finalize_mean(float* __restrict__ out,
                                                     const float* __restrict__ counts, int total) {
    int i = blockIdx.x * 256 + threadIdx.x;
    if (i < total) {
        float c = counts[i / 28];
        out[i] *= (1.0f / fmaxf(c, 1.0f));
    }
}

// ---- fused: gather -> GEMM1(WMMA) -> relu -> GEMM2(WMMA) -> tensor product -> atomic scatter ----
__global__ __launch_bounds__(TPB) void tpconv(const float* __restrict__ node_attr,
                                              const int*   __restrict__ eidx,      // [2,E]
                                              const float* __restrict__ edge_attr, // [E,48]
                                              const float* __restrict__ edge_sh,   // [E,9]
                                              const float* __restrict__ b1,
                                              const float* __restrict__ b2,
                                              const u32*   __restrict__ swgt_g,
                                              float* __restrict__ out,
                                              float* __restrict__ counts,
                                              int E) {
    __shared__ __align__(16) u32   swgt[SWGT_DW];
    __shared__ __align__(16) u32   s_ea[WPB][ETILE * KPAD / 2];  // edge_attr f16, later reused for h
    __shared__ __align__(16) float s_x[WPB][ETILE * NSC];
    __shared__ __align__(16) float s_sh[WPB][ETILE * 12];
    __shared__ int s_seg[WPB][ETILE];
    __shared__ int s_dst[WPB][ETILE];

    const int tid  = threadIdx.x;
    const int wid  = tid >> 5;
    const int lane = tid & 31;
    const int row  = lane & 15;
    const int hi   = lane >> 4;

    // --- stage pre-swizzled weights into LDS (async-to-LDS path when available) ---
#if __has_builtin(__builtin_amdgcn_global_load_async_to_lds_b128) && __has_builtin(__builtin_amdgcn_s_wait_asynccnt)
    {
        typedef __attribute__((address_space(1))) vi4* gp_t;
        typedef __attribute__((address_space(3))) vi4* lp_t;
        for (int i = tid; i < SWGT_DW / 4; i += TPB)
            __builtin_amdgcn_global_load_async_to_lds_b128((gp_t)(swgt_g + i * 4),
                                                           (lp_t)(swgt + i * 4), 0, 0);
        __builtin_amdgcn_s_wait_asynccnt(0);
    }
#else
    for (int i = tid; i < SWGT_DW / 4; i += TPB)
        ((vi4*)swgt)[i] = ((const vi4*)swgt_g)[i];
#endif

    const int tile = blockIdx.x * WPB + wid;
    const int e0r  = tile * ETILE;   // real first edge of this wave's tile

    u32*   ea  = s_ea[wid];
    f16*   eah = (f16*)ea;
    float* xw  = s_x[wid];
    float* shw = s_sh[wid];

    // zero K-pad region (halfs 48..63 -> dwords 24..31 of each 32-dword row)
    for (int idx = lane; idx < ETILE * 8; idx += 32) {
        int e = idx >> 3, d = idx & 7;
        ea[e * 32 + 24 + d] = 0u;
    }
    // edge_attr tile -> f16 (coalesced; per-edge clamp for tail safety)
    for (int idx = lane; idx < ETILE * 48; idx += 32) {
        int e = idx / 48, k = idx - e * 48;
        int g = e0r + e; if (g >= E) g = E - 1; if (g < 0) g = 0;
        eah[e * KPAD + k] = (f16)edge_attr[(size_t)g * 48 + k];
    }
    if (lane < ETILE) {
        int g = e0r + lane; if (g >= E) g = E - 1; if (g < 0) g = 0;
        s_seg[wid][lane] = eidx[g];
        s_dst[wid][lane] = eidx[E + g];
    }
    for (int idx = lane; idx < ETILE * 9; idx += 32) {
        int e = idx / 9, c = idx - e * 9;
        int g = e0r + e; if (g >= E) g = E - 1; if (g < 0) g = 0;
        shw[e * 12 + c] = edge_sh[(size_t)g * 9 + c];
    }
    __syncthreads();
    // gather destination-node features
    for (int idx = lane; idx < ETILE * NSC; idx += 32) {
        int e = idx >> 4, i = idx & 15;
        xw[idx] = node_attr[(size_t)s_dst[wid][e] * NSC + i];
    }
    __syncthreads();

    // --- GEMM1 A fragments (16x32 f16 per K-step) ---
    v16h a0, a1;
    {
        const u32* base = ea + row * 32 + hi * 4;
        a0 = ldA(base);       // K = 0..31
        a1 = ldA(base + 16);  // K = 32..63 (zero-padded beyond 48)
    }
    // --- GEMM1: h = relu(ea @ w1 + b1); write f16 h back over s_ea ---
    for (int nt = 0; nt < 3; ++nt) {
        v8f c = {};
        const u32* bcol = swgt + (nt * 16 + row) * 32 + hi * 4;
        v16h b = ldB(bcol);
        c = __builtin_amdgcn_wmma_f32_16x16x32_f16(false, a0, false, b, (short)0, c, false, false);
        b = ldB(bcol + 16);
        c = __builtin_amdgcn_wmma_f32_16x16x32_f16(false, a1, false, b, (short)0, c, false, false);
        float bias = b1[nt * 16 + row];
        for (int r = 0; r < 8; ++r) {
            float v = c[r] + bias;
            v = v > 0.0f ? v : 0.0f;
            eah[(r + hi * 8) * KPAD + nt * 16 + row] = (f16)v;  // transpose D -> row-major h
        }
    }
    __syncthreads();

    // --- GEMM2 A fragments from h ---
    v16h h0, h1;
    {
        const u32* base = ea + row * 32 + hi * 4;
        h0 = ldA(base);
        h1 = ldA(base + 16);
    }

    // --- GEMM2 fused with tensor product on D fragments ---
    // part 1: features 0..255 = w00[e][i=nt2][j]; accumulate out0 in C/D layout
    float acc0[8];
    for (int r = 0; r < 8; ++r) acc0[r] = 0.0f;
#pragma unroll 4
    for (int nt2 = 0; nt2 < 16; ++nt2) {
        v8f c = {};
        const u32* bcol = swgt + W2_OFF_DW + (nt2 * 16 + row) * 32 + hi * 4;
        v16h b = ldB(bcol);
        c = __builtin_amdgcn_wmma_f32_16x16x32_f16(false, h0, false, b, (short)0, c, false, false);
        b = ldB(bcol + 16);
        c = __builtin_amdgcn_wmma_f32_16x16x32_f16(false, h1, false, b, (short)0, c, false, false);
        float bias = b2[nt2 * 16 + row];
        for (int r = 0; r < 8; ++r) {
            int e = r + hi * 8;
            acc0[r] += (xw[e * NSC + nt2] * shw[e * 12]) * (c[r] + bias);
        }
    }
    // part 2: features 256..319 = w01[e][i][u]; this lane holds u=row&3, i=(nt2-16)*4 + row>>2
    float acc1[8];
    for (int r = 0; r < 8; ++r) acc1[r] = 0.0f;
#pragma unroll 4
    for (int nt2 = 16; nt2 < 20; ++nt2) {
        v8f c = {};
        const u32* bcol = swgt + W2_OFF_DW + (nt2 * 16 + row) * 32 + hi * 4;
        v16h b = ldB(bcol);
        c = __builtin_amdgcn_wmma_f32_16x16x32_f16(false, h0, false, b, (short)0, c, false, false);
        b = ldB(bcol + 16);
        c = __builtin_amdgcn_wmma_f32_16x16x32_f16(false, h1, false, b, (short)0, c, false, false);
        float bias = b2[nt2 * 16 + row];
        int i = (nt2 - 16) * 4 + (row >> 2);
        for (int r = 0; r < 8; ++r) {
            int e = r + hi * 8;
            acc1[r] += xw[e * NSC + i] * (c[r] + bias);
        }
    }

    // --- scatter-add (mean finalized later) ---
    const float inv = 0.25f;          // 1/sqrt(16)
    const int u = row & 3;            // vector-channel index
    const int m = row >> 2;           // 0..2 -> SH component writer, 3 -> count lane
    for (int r = 0; r < 8; ++r) {
        int e = r + hi * 8;
        int ge = e0r + e;
        float s = acc1[r];
        s += __shfl_xor(s, 4);        // reduce over i-groups (lanes differing in bits 2..3)
        s += __shfl_xor(s, 8);
        if (ge < E) {
            int seg = s_seg[wid][e];
            atomAdd(&out[(size_t)seg * 28 + row], acc0[r] * inv);
            if (m < 3)
                atomAdd(&out[(size_t)seg * 28 + 16 + u * 3 + m], s * shw[e * 12 + 1 + m] * inv);
            else if (u == 3)
                atomAdd(&counts[seg], 1.0f);
        }
    }
}

extern "C" void kernel_launch(void* const* d_in, const int* in_sizes, int n_in,
                              void* d_out, int out_size, void* d_ws, size_t ws_size,
                              hipStream_t stream) {
    const float* node_attr = (const float*)d_in[0];
    const int*   eidx      = (const int*)d_in[1];
    const float* edge_attr = (const float*)d_in[2];
    const float* edge_sh   = (const float*)d_in[3];
    const float* w1        = (const float*)d_in[4];
    const float* b1        = (const float*)d_in[5];
    const float* w2        = (const float*)d_in[6];
    const float* b2        = (const float*)d_in[7];
    const int E = in_sizes[2] / 48;
    const int N = in_sizes[0] / 16;
    float* out = (float*)d_out;

    u32*   swgt_g = (u32*)d_ws;
    float* cnts   = (float*)d_ws + 12288;   // counts after weight blob (aligned region)

    prep_weights<<<(SWGT_DW + 255) / 256, 256, 0, stream>>>(w1, w2, swgt_g);
    const int total = N * 28;
    zero_f32<<<(total + 255) / 256, 256, 0, stream>>>(out, total);
    zero_f32<<<(N + 255) / 256, 256, 0, stream>>>(cnts, N);

    const int ntiles = (E + ETILE - 1) / ETILE;
    const int blocks = (ntiles + WPB - 1) / WPB;
    tpconv<<<blocks, TPB, 0, stream>>>(node_attr, eidx, edge_attr, edge_sh,
                                       b1, b2, swgt_g, out, cnts, E);

    finalize_mean<<<(total + 255) / 256, 256, 0, stream>>>(out, cnts, total);
}
